// EquivariantNetwork_51891794870657
// MI455X (gfx1250) — matrix operand order repaired and checked by reading, not compiled, not run on hardware
//
#include <hip/hip_runtime.h>
#include <hip/hip_bf16.h>
#include <math.h>

typedef __attribute__((ext_vector_type(16))) _Float16 v16h;
typedef __attribute__((ext_vector_type(8)))  float    v8f;

#define N_NODES   25000
#define N_EDGES   400000
#define F_IN      8
#define NB_R      8
#define S0_W      64
#define V0_W      32
#define MS_W      72        // F + S0
#define NW_BIG    240       // 2*MS + 3*MV
#define AGG_S_W   104       // MS + MV
#define AGG_V_W   136       // MS + 2*MV
#define INV_DEG   0.25f     // (E/N)^-0.5 = 16^-0.5
#define SQRT3_C   1.7320508075688772f
#define INV_SQRT3 0.5773502691896258f
#define INV_SQRT2 0.7071067811865476f

// fast silu: hardware v_rcp_f32 instead of IEEE div expansion
__device__ __forceinline__ float silu_f(float x)
{
    return x * __builtin_amdgcn_rcpf(1.0f + __expf(-x));
}

// ---------------------------------------------------------------------------
// Pack a (K x Ncols) f32 weight matrix into WMMA B-fragment order (f16),
// padded to 32-K steps.  Fragment layout:  frag[(tile*ksteps+ks)*32 + lane]
// is a v16h where element h = B[k][n],  n = tile*16 + (lane&15),
// k = ks*32 + ((lane&16)?16:0) + h.
// ---------------------------------------------------------------------------
__global__ void pack_b_frags(const float* __restrict__ W, _Float16* __restrict__ out,
                             int K, int Ncols, int ntiles, int ksteps)
{
    int idx = blockIdx.x * blockDim.x + threadIdx.x;
    int total = ntiles * ksteps * 32 * 16;
    if (idx >= total) return;
    int h    = idx & 15;
    int lane = (idx >> 4) & 31;
    int rest = idx >> 9;              // = tile*ksteps + ks
    int ks   = rest % ksteps;
    int tile = rest / ksteps;
    int n = tile * 16 + (lane & 15);
    int k = ks * 32 + ((lane & 16) ? 16 : 0) + h;
    out[idx] = (k < K) ? (_Float16)W[k * Ncols + n] : (_Float16)0.0f;
}

// ---------------------------------------------------------------------------
// Per-edge geometry: r, sh1 = sqrt(3)*vec/r, radial embedding (f16, 8 vals)
// ---------------------------------------------------------------------------
__global__ void edge_prep(const float* __restrict__ ev,
                          float* __restrict__ sh1, _Float16* __restrict__ remb)
{
    int e = blockIdx.x * blockDim.x + threadIdx.x;
    if (e >= N_EDGES) return;
    float x = ev[e * 3 + 0], y = ev[e * 3 + 1], z = ev[e * 3 + 2];
    float r = sqrtf(x * x + y * y + z * z);
    float rinv = __builtin_amdgcn_rcpf(r);
    sh1[e * 3 + 0] = SQRT3_C * x * rinv;
    sh1[e * 3 + 1] = SQRT3_C * y * rinv;
    sh1[e * 3 + 2] = SQRT3_C * z * rinv;
    float xr = r * 0.25f;                       // r / R_MAX
    float env = 0.0f;
    if (xr < 1.0f) {
        float x2 = xr * xr, x3 = x2 * xr, x6 = x3 * x3, x7 = x6 * xr, x8 = x7 * xr;
        env = 1.0f - 28.0f * x6 + 48.0f * x7 - 21.0f * x8;   // p = 6 envelope
    }
    float base = r * 0.7853981633974483f;       // pi * r / R_MAX
    float c = 0.7071067811865476f * rinv * env; // sqrt(2/R_MAX) * env / r
    union { _Float16 h[8]; uint4 u; } o;
#pragma unroll
    for (int n = 1; n <= 8; ++n) o.h[n - 1] = (_Float16)(c * __sinf((float)n * base));
    *(uint4*)(remb + (size_t)e * NB_R) = o.u;
}

__global__ void fill_zero(float* __restrict__ p, int count)
{
    for (int i = blockIdx.x * blockDim.x + threadIdx.x; i < count;
         i += gridDim.x * blockDim.x) p[i] = 0.0f;
}

// ---------------------------------------------------------------------------
// One wave computes the radial MLP for 16 edges via WMMA:
//   H = silu(remb @ W1 + b1)  (16x64),   Wout = H @ W2 + b2  (16 x NT*16)
// Result lands (f16) in this wave's wtile LDS slice, row = edge-in-tile.
// ---------------------------------------------------------------------------
template <int NT>
__device__ __forceinline__ void edge_mlp_wave(
    int lane, int ebase,
    const _Float16* __restrict__ remb,
    const _Float16* __restrict__ w1f, const float* __restrict__ b1,
    const _Float16* __restrict__ w2f, const float* __restrict__ b2,
    _Float16 (&hbuf)[16][64], _Float16 (&wtile)[16][NT * 16])
{
    union AF { v16h v; _Float16 h[16]; };
    union CF { v8f  v; float    f[8];  };
    const int col = lane & 15;
    const int hi  = lane >> 4;

    // ---- A fragment: 16 edges x K32 (r_emb in K 0..7, rest zero) ----
    AF a1;
#pragma unroll
    for (int i = 0; i < 16; ++i) a1.h[i] = (_Float16)0.0f;
    if (hi == 0) {
        const _Float16* re = remb + (size_t)(ebase + col) * NB_R;
#pragma unroll
        for (int i = 0; i < NB_R; ++i) a1.h[i] = re[i];
    }

    // ---- GEMM1: 4 N-tiles of W1 (K padded 8->32), bias + silu -> hbuf ----
    const v16h* b1f = (const v16h*)w1f;
#pragma unroll
    for (int t = 0; t < 4; ++t) {
        CF c;
        float bias = b1[t * 16 + col];
#pragma unroll
        for (int r = 0; r < 8; ++r) c.f[r] = bias;
        v16h bfrag = b1f[t * 32 + lane];
        c.v = __builtin_amdgcn_wmma_f32_16x16x32_f16(false, a1.v, false, bfrag,
                                                     (short)0, c.v, false, false);
#pragma unroll
        for (int r = 0; r < 8; ++r)
            hbuf[r + hi * 8][t * 16 + col] = (_Float16)silu_f(c.f[r]);
    }
    __syncthreads();

    // ---- re-load H as A fragments (two K32 steps) ----
    AF a2[2];
#pragma unroll
    for (int ks = 0; ks < 2; ++ks) {
#pragma unroll
        for (int h = 0; h < 16; ++h) {
            int k = ks * 32 + hi * 16 + h;
            a2[ks].h[h] = hbuf[col][k];
        }
    }

    // ---- GEMM2: NT N-tiles of W2, K=64 -> wtile ----
    const v16h* b2f = (const v16h*)w2f;
#pragma unroll
    for (int t = 0; t < NT; ++t) {
        CF c;
        float bias = b2[t * 16 + col];
#pragma unroll
        for (int r = 0; r < 8; ++r) c.f[r] = bias;
        c.v = __builtin_amdgcn_wmma_f32_16x16x32_f16(false, a2[0].v, false,
                 b2f[(t * 2 + 0) * 32 + lane], (short)0, c.v, false, false);
        c.v = __builtin_amdgcn_wmma_f32_16x16x32_f16(false, a2[1].v, false,
                 b2f[(t * 2 + 1) * 32 + lane], (short)0, c.v, false, false);
#pragma unroll
        for (int r = 0; r < 8; ++r)
            wtile[r + hi * 8][t * 16 + col] = (_Float16)c.f[r];
    }
    __syncthreads();
}

// ---------------------------------------------------------------------------
// Layer 0 fused: MLP0 (8->64->16) + scatter messages (2 lanes / edge)
// ---------------------------------------------------------------------------
__global__ void __launch_bounds__(128)
fused_layer0(const _Float16* __restrict__ remb,
             const _Float16* __restrict__ w1f, const float* __restrict__ b1,
             const _Float16* __restrict__ w2f, const float* __restrict__ b2,
             const int* __restrict__ eidx, const float* __restrict__ sh1,
             const float* __restrict__ x_feat,
             float* __restrict__ agg_s, float* __restrict__ agg_v)
{
    __shared__ _Float16 hbuf[4][16][64];
    __shared__ _Float16 wtile[4][16][16];
    int lane = threadIdx.x & 31, wid = threadIdx.x >> 5;
    int ebase = blockIdx.x * 64 + wid * 16;

    int el = lane & 15, half = lane >> 4;
    int e = ebase + el;
    int src = eidx[e], dst = eidx[N_EDGES + e];
    const float* xf = x_feat + (size_t)src * F_IN;
    __builtin_prefetch(xf, 0, 0);

    edge_mlp_wave<1>(lane, ebase, remb, w1f, b1, w2f, b2, hbuf[wid], wtile[wid]);

    float s0 = sh1[e * 3 + 0], s1 = sh1[e * 3 + 1], s2 = sh1[e * 3 + 2];
    const _Float16* w = &wtile[wid][el][0];
    float* as = agg_s + (size_t)dst * AGG_S_W;
    float* av = agg_v + (size_t)dst * (AGG_V_W * 3);
#pragma unroll
    for (int i = 0; i < 4; ++i) {
        int f = half * 4 + i;
        float sv  = xf[f];
        atomicAdd(as + f, sv * (float)w[f] * INV_DEG);
        float m = sv * (float)w[F_IN + f] * INV_DEG;
        atomicAdd(av + f * 3 + 0, m * s0);
        atomicAdd(av + f * 3 + 1, m * s1);
        atomicAdd(av + f * 3 + 2, m * s2);
    }
}

// ---------------------------------------------------------------------------
// Layers 1/2 fused: MLP (8->64->240) + gather s_in/v + scatter messages
// ---------------------------------------------------------------------------
__global__ void __launch_bounds__(128)
fused_layer12(const _Float16* __restrict__ remb,
              const _Float16* __restrict__ w1f, const float* __restrict__ b1,
              const _Float16* __restrict__ w2f, const float* __restrict__ b2,
              const int* __restrict__ eidx, const float* __restrict__ sh1,
              const float* __restrict__ sbuf, const float* __restrict__ vbuf,
              float* __restrict__ agg_s, float* __restrict__ agg_v)
{
    __shared__ _Float16 hbuf[4][16][64];
    __shared__ _Float16 wtile[4][16][NW_BIG];
    int lane = threadIdx.x & 31, wid = threadIdx.x >> 5;
    int ebase = blockIdx.x * 64 + wid * 16;

    int el = lane & 15, half = lane >> 4;
    int e = ebase + el;
    int src = eidx[e], dst = eidx[N_EDGES + e];
    const float* sb = sbuf + (size_t)src * MS_W;
    const float* vb = vbuf + (size_t)src * (V0_W * 3);
    __builtin_prefetch(sb + half * 36, 0, 0);
    __builtin_prefetch(vb + half * 48, 0, 0);

    edge_mlp_wave<15>(lane, ebase, remb, w1f, b1, w2f, b2, hbuf[wid], wtile[wid]);

    float s0 = sh1[e * 3 + 0], s1 = sh1[e * 3 + 1], s2 = sh1[e * 3 + 2];
    const _Float16* w = &wtile[wid][el][0];
    float* as = agg_s + (size_t)dst * AGG_S_W;
    float* av = agg_v + (size_t)dst * (AGG_V_W * 3);

    // scalar channels: s_e * w_ss  and  s_e (x) sh1 * w_sv     (72 split 36/36)
    for (int f = half * 36; f < half * 36 + 36; ++f) {
        float sv = sb[f];
        atomicAdd(as + f, sv * (float)w[f] * INV_DEG);
        float m = sv * (float)w[104 + f] * INV_DEG;           // w_sv
        atomicAdd(av + f * 3 + 0, m * s0);
        atomicAdd(av + f * 3 + 1, m * s1);
        atomicAdd(av + f * 3 + 2, m * s2);
    }
    // vector channels: dot, v*w_v0, cross*w_v1                (32 split 16/16)
    for (int o = half * 16; o < half * 16 + 16; ++o) {
        float v0 = vb[o * 3 + 0], v1 = vb[o * 3 + 1], v2 = vb[o * 3 + 2];
        float dot = (v0 * s0 + v1 * s1 + v2 * s2) * INV_SQRT3;
        atomicAdd(as + 72 + o, dot * (float)w[72 + o] * INV_DEG);   // w_vs
        float w0 = (float)w[176 + o] * INV_DEG;                      // w_v0
        atomicAdd(av + (72 + o) * 3 + 0, v0 * w0);
        atomicAdd(av + (72 + o) * 3 + 1, v1 * w0);
        atomicAdd(av + (72 + o) * 3 + 2, v2 * w0);
        float c0 = v1 * s2 - v2 * s1;
        float c1 = v2 * s0 - v0 * s2;
        float c2 = v0 * s1 - v1 * s0;
        float w1v = (float)w[208 + o] * (INV_DEG * INV_SQRT2);       // w_v1
        atomicAdd(av + (104 + o) * 3 + 0, c0 * w1v);
        atomicAdd(av + (104 + o) * 3 + 1, c1 * w1v);
        atomicAdd(av + (104 + o) * 3 + 2, c2 * w1v);
    }
}

// ---------------------------------------------------------------------------
// Node update after layer 0:  s = silu(agg_s @ lin0_ws),  v = agg_v @ lin0_wv
// Also seeds sbuf[:, 0:8] with x_feat (node_copy).
// ---------------------------------------------------------------------------
__global__ void __launch_bounds__(64)
node0(const float* __restrict__ agg_s, const float* __restrict__ agg_v,
      const float* __restrict__ lws, const float* __restrict__ lwv,
      const float* __restrict__ x_feat,
      float* __restrict__ sbuf, float* __restrict__ vbuf)
{
    int n = blockIdx.x, t = threadIdx.x;
    __shared__ float as[8], av[24];
    if (t < 8)  as[t] = agg_s[(size_t)n * AGG_S_W + t];
    if (t < 24) av[t] = agg_v[(size_t)n * (AGG_V_W * 3) + t];
    __syncthreads();
    float acc = 0.0f;
#pragma unroll
    for (int m = 0; m < 8; ++m) acc += as[m] * lws[m * 64 + t];
    sbuf[(size_t)n * MS_W + F_IN + t] = silu_f(acc);
    if (t < 8) sbuf[(size_t)n * MS_W + t] = x_feat[(size_t)n * F_IN + t];
    if (t < 32) {
#pragma unroll
        for (int d = 0; d < 3; ++d) {
            float a = 0.0f;
#pragma unroll
            for (int m = 0; m < 8; ++m) a += av[m * 3 + d] * lwv[m * 32 + t];
            vbuf[((size_t)n * V0_W + t) * 3 + d] = a;
        }
    }
}

// ---------------------------------------------------------------------------
// Node update layers 1/2:  s += silu(agg_s @ ls),  v += agg_v @ lv
// ---------------------------------------------------------------------------
__global__ void __launch_bounds__(64)
node12(const float* __restrict__ agg_s, const float* __restrict__ agg_v,
       const float* __restrict__ lws, const float* __restrict__ lwv,
       float* __restrict__ sbuf, float* __restrict__ vbuf)
{
    int n = blockIdx.x, t = threadIdx.x;
    __shared__ float as[AGG_S_W], av[AGG_V_W * 3];
    for (int i = t; i < AGG_S_W; i += 64)     as[i] = agg_s[(size_t)n * AGG_S_W + i];
    for (int i = t; i < AGG_V_W * 3; i += 64) av[i] = agg_v[(size_t)n * (AGG_V_W * 3) + i];
    __syncthreads();
    float acc = 0.0f;
    for (int m = 0; m < AGG_S_W; ++m) acc += as[m] * lws[m * 64 + t];
    sbuf[(size_t)n * MS_W + F_IN + t] += silu_f(acc);
    if (t < 32) {
#pragma unroll
        for (int d = 0; d < 3; ++d) {
            float a = 0.0f;
            for (int m = 0; m < AGG_V_W; ++m) a += av[m * 3 + d] * lwv[m * 32 + t];
            vbuf[((size_t)n * V0_W + t) * 3 + d] += a;
        }
    }
}

// ---------------------------------------------------------------------------
// Readout: feats = [s(64), |v|(32)] ;  out = feats @ ro_w[ptype] + ro_b[ptype]
// ---------------------------------------------------------------------------
__global__ void __launch_bounds__(64)
readout(const float* __restrict__ sbuf, const float* __restrict__ vbuf,
        const int* __restrict__ ptype, const float* __restrict__ ro_w,
        const float* __restrict__ ro_b, float* __restrict__ out)
{
    int n = blockIdx.x, t = threadIdx.x;
    __shared__ float feats[96];
    feats[t] = sbuf[(size_t)n * MS_W + F_IN + t];
    if (t < 32) {
        float v0 = vbuf[((size_t)n * V0_W + t) * 3 + 0];
        float v1 = vbuf[((size_t)n * V0_W + t) * 3 + 1];
        float v2 = vbuf[((size_t)n * V0_W + t) * 3 + 2];
        feats[64 + t] = sqrtf(v0 * v0 + v1 * v1 + v2 * v2 + 1e-12f);
    }
    __syncthreads();
    if (t < 10) {
        int p = ptype[n];
        float acc = ro_b[p * 10 + t];
        for (int k = 0; k < 96; ++k) acc += feats[k] * ro_w[p * 960 + k * 10 + t];
        out[(size_t)n * 10 + t] = acc;
    }
}

// ---------------------------------------------------------------------------
extern "C" void kernel_launch(void* const* d_in, const int* in_sizes, int n_in,
                              void* d_out, int out_size, void* d_ws, size_t ws_size,
                              hipStream_t stream)
{
    (void)in_sizes; (void)n_in; (void)out_size; (void)ws_size;
    const float* x_feat   = (const float*)d_in[0];
    const int*   ptype    = (const int*)d_in[1];
    const int*   eidx     = (const int*)d_in[2];
    const float* edge_vec = (const float*)d_in[3];
    const float* mlp_w1[3] = { (const float*)d_in[4],  (const float*)d_in[10], (const float*)d_in[16] };
    const float* mlp_b1[3] = { (const float*)d_in[5],  (const float*)d_in[11], (const float*)d_in[17] };
    const float* mlp_w2[3] = { (const float*)d_in[6],  (const float*)d_in[12], (const float*)d_in[18] };
    const float* mlp_b2[3] = { (const float*)d_in[7],  (const float*)d_in[13], (const float*)d_in[19] };
    const float* lin_ws[3] = { (const float*)d_in[8],  (const float*)d_in[14], (const float*)d_in[20] };
    const float* lin_wv[3] = { (const float*)d_in[9],  (const float*)d_in[15], (const float*)d_in[21] };
    const float* ro_w = (const float*)d_in[22];
    const float* ro_b = (const float*)d_in[23];
    float* out = (float*)d_out;

    // ---- workspace bump allocator (~80 MB total) ----
    char* ws = (char*)d_ws;
    size_t off = 0;
    auto alloc = [&](size_t bytes) -> char* {
        char* p = ws + off;
        off = (off + bytes + 255) & ~(size_t)255;
        return p;
    };
    float*    sh1  = (float*)   alloc((size_t)N_EDGES * 3 * sizeof(float));
    _Float16* remb = (_Float16*)alloc((size_t)N_EDGES * NB_R * sizeof(_Float16));
    float*    sbuf = (float*)   alloc((size_t)N_NODES * MS_W * sizeof(float));
    float*    vbuf = (float*)   alloc((size_t)N_NODES * V0_W * 3 * sizeof(float));
    float*    agg_s = (float*)  alloc((size_t)N_NODES * (AGG_S_W + AGG_V_W * 3) * sizeof(float));
    float*    agg_v = agg_s + (size_t)N_NODES * AGG_S_W;
    _Float16* w1f[3]; _Float16* w2f[3];
    for (int i = 0; i < 3; ++i) w1f[i] = (_Float16*)alloc(4 * 1 * 512 * sizeof(_Float16));
    w2f[0] = (_Float16*)alloc(1 * 2 * 512 * sizeof(_Float16));
    w2f[1] = (_Float16*)alloc(15 * 2 * 512 * sizeof(_Float16));
    w2f[2] = (_Float16*)alloc(15 * 2 * 512 * sizeof(_Float16));

    // ---- pack MLP weights into WMMA B-fragment order (f16) ----
    for (int i = 0; i < 3; ++i)
        pack_b_frags<<<(2048 + 255) / 256, 256, 0, stream>>>(mlp_w1[i], w1f[i], 8, 64, 4, 1);
    pack_b_frags<<<(1024 + 255) / 256, 256, 0, stream>>>(mlp_w2[0], w2f[0], 64, 16, 1, 2);
    pack_b_frags<<<(15360 + 255) / 256, 256, 0, stream>>>(mlp_w2[1], w2f[1], 64, 240, 15, 2);
    pack_b_frags<<<(15360 + 255) / 256, 256, 0, stream>>>(mlp_w2[2], w2f[2], 64, 240, 15, 2);

    edge_prep<<<(N_EDGES + 255) / 256, 256, 0, stream>>>(edge_vec, sh1, remb);

    const int agg_cnt = N_NODES * (AGG_S_W + AGG_V_W * 3);
    const int eblocks = N_EDGES / 64;   // exact: 6250

    // ---- layer 0 ----
    fill_zero<<<2048, 256, 0, stream>>>(agg_s, agg_cnt);
    fused_layer0<<<eblocks, 128, 0, stream>>>(remb, w1f[0], mlp_b1[0], w2f[0], mlp_b2[0],
                                              eidx, sh1, x_feat, agg_s, agg_v);
    node0<<<N_NODES, 64, 0, stream>>>(agg_s, agg_v, lin_ws[0], lin_wv[0], x_feat, sbuf, vbuf);

    // ---- layers 1 and 2 ----
    for (int L = 1; L <= 2; ++L) {
        fill_zero<<<2048, 256, 0, stream>>>(agg_s, agg_cnt);
        fused_layer12<<<eblocks, 128, 0, stream>>>(remb, w1f[L], mlp_b1[L], w2f[L], mlp_b2[L],
                                                   eidx, sh1, sbuf, vbuf, agg_s, agg_v);
        node12<<<N_NODES, 64, 0, stream>>>(agg_s, agg_v, lin_ws[L], lin_wv[L], sbuf, vbuf);
    }

    // ---- readout ----
    readout<<<N_NODES, 64, 0, stream>>>(sbuf, vbuf, ptype, ro_w, ro_b, out);
}
